// MultiHeadAttention_12532714570261
// MI455X (gfx1250) — matrix operand order repaired and checked by reading, not compiled, not run on hardware
//
#include <hip/hip_runtime.h>

// ---------------------------------------------------------------------------
// MHA block for MI455X (gfx1250): f16 WMMA (16x16x32, f32 accum) for all four
// GEMM stages; f32 softmax/LayerNorm. wave32 throughout.
// Weights pre-converted to f16 and transposed so every B-fragment is a
// contiguous 32-byte per-lane load (no per-WMMA cvt, no strided b32 loads).
// ---------------------------------------------------------------------------

typedef __attribute__((ext_vector_type(16))) _Float16 v16h;
typedef __attribute__((ext_vector_type(8)))  float    v8f;

#define N_HEAD  8
#define D_MODEL 512
#define D_HEAD  64
#define BATCH   16
#define SEQ     1024
#define M_ROWS  (BATCH * SEQ)                     // 16384
#define PROJ_ELEMS ((size_t)M_ROWS * D_MODEL)     // 8388608
#define W_ELEMS    ((size_t)D_MODEL * D_MODEL)    // 262144
#define OUT_ELEMS  PROJ_ELEMS                     // first output ("out") size

static __device__ __forceinline__ v8f wmma16(v16h a, v16h b, v8f c) {
    // v_wmma_f32_16x16x32_f16  (neg_a, A, neg_b, B, c_mod, C, reuse_a, reuse_b)
    return __builtin_amdgcn_wmma_f32_16x16x32_f16(false, a, false, b,
                                                  (short)0, c, false, false);
}

// ---------------------------------------------------------------------------
// Kernel 0: convert + transpose the 4 weight matrices to f16 W^T [N, K].
// 1M elements total -> negligible one-time cost; weights then live in L2.
// ---------------------------------------------------------------------------
__global__ __launch_bounds__(256)
void prep_weights_kernel(const float* __restrict__ Wq, const float* __restrict__ Wk,
                         const float* __restrict__ Wv, const float* __restrict__ Wfc,
                         _Float16* __restrict__ WqT, _Float16* __restrict__ WkT,
                         _Float16* __restrict__ WvT, _Float16* __restrict__ WfcT) {
    const int id = blockIdx.x * 256 + threadIdx.x;    // 4 * 512 * 512 threads
    const int w   = id >> 18;
    const int rem = id & 262143;
    const int kk  = rem >> 9;       // source row
    const int nn  = rem & 511;      // source col (coalesced across threads)
    const float* W  = (w == 0) ? Wq  : (w == 1) ? Wk  : (w == 2) ? Wv  : Wfc;
    _Float16*   WT  = (w == 0) ? WqT : (w == 1) ? WkT : (w == 2) ? WvT : WfcT;
    WT[(size_t)nn * D_MODEL + kk] = (_Float16)W[(size_t)kk * D_MODEL + nn];
}

// ---------------------------------------------------------------------------
// Kernel A: fused Q/K/V projection GEMMs.
// One wave computes a 16(M) x 64(N) strip: 4 accumulators share each A-frag.
// Q,K outputs: f16 [B, H, L, 64].  V output: f16 TRANSPOSED [B, H, 64, L]
// so the attention P@V B-fragments are contiguous per lane.
// ---------------------------------------------------------------------------
__global__ __launch_bounds__(256)
void proj_gemm_kernel(const float* __restrict__ qin, const float* __restrict__ kin,
                      const float* __restrict__ vin,
                      const _Float16* __restrict__ WqT, const _Float16* __restrict__ WkT,
                      const _Float16* __restrict__ WvT,
                      _Float16* __restrict__ qh, _Float16* __restrict__ kh,
                      _Float16* __restrict__ vhT) {
    const int gid  = blockIdx.x * 8 + (threadIdx.x >> 5);   // wave id
    const int lane = threadIdx.x & 31;
    const int mat  = gid >> 13;          // 0:q 1:k 2:v   (8192 waves per matrix)
    const int rem  = gid & 8191;
    const int mt   = rem >> 3;           // 0..1023  (16-row tiles)
    const int n64  = rem & 7;            // 0..7     (64-col strips)

    const float*    A  = (mat == 0) ? qin : (mat == 1) ? kin : vin;
    const _Float16* WT = (mat == 0) ? WqT : (mat == 1) ? WkT : WvT;

    const int hg = lane >> 4;            // half-wave group
    const int cn = lane & 15;
    const int kb = hg * 16;

    // A fragment source: row = mt*16 + cn, K pattern {kb..kb+15}
    const float* Arow = A + (size_t)(mt * 16 + cn) * D_MODEL + kb;

    v8f acc[4] = {};
#pragma unroll 1
    for (int kc = 0; kc < 16; ++kc) {    // K = 512 in chunks of 32
        const float* pa = Arow + kc * 32;
        __builtin_prefetch(Arow + (kc + 1) * 32, 0, 0);
        v16h a;
#pragma unroll
        for (int j = 0; j < 8; ++j) {
            a[j]     = (_Float16)pa[j];
            a[8 + j] = (_Float16)pa[8 + j];
        }
#pragma unroll
        for (int s = 0; s < 4; ++s) {
            // B frag from W^T: lane n = n64*64+s*16+cn, 16 contiguous f16 in K
            const _Float16* pb = WT + (size_t)(n64 * 64 + s * 16 + cn) * D_MODEL
                                    + kc * 32 + hg * 16;
            v16h b;
#pragma unroll
            for (int e = 0; e < 16; ++e) b[e] = pb[e];
            acc[s] = wmma16(a, b, acc[s]);
        }
    }

    // store
#pragma unroll
    for (int s = 0; s < 4; ++s) {
#pragma unroll
        for (int i = 0; i < 8; ++i) {
            int row = mt * 16 + i + 8 * hg;          // global M row
            int bb = row >> 10, ll = row & 1023;
            int d  = s * 16 + cn;                    // head-local dim (h == n64)
            _Float16 val = (_Float16)acc[s][i];
            if (mat == 2) {
                // V transposed: [B, H, 64, L]
                vhT[(((size_t)bb * N_HEAD + n64) * D_HEAD + d) * SEQ + ll] = val;
            } else {
                _Float16* O = (mat == 0) ? qh : kh;  // [B, H, L, 64]
                O[(((size_t)bb * N_HEAD + n64) * SEQ + ll) * D_HEAD + d] = val;
            }
        }
    }
}

// ---------------------------------------------------------------------------
// Kernel B: attention for one (b, h, 16-query-row tile) per wave.
// Scores (f32) parked in 64KB LDS; exact two-pass softmax; probs written f32
// to the attn output and repacked in-place to f16 for the P@V WMMAs.
// ---------------------------------------------------------------------------
__global__ __launch_bounds__(32)
void attn_kernel(const _Float16* __restrict__ qh, const _Float16* __restrict__ kh,
                 const _Float16* __restrict__ vhT, const unsigned char* __restrict__ mask,
                 float* __restrict__ attn_out, _Float16* __restrict__ o_out) {
    __shared__ float S[16 * SEQ];                     // 64 KB (of 320 KB/WGP)

    const int lane = threadIdx.x;
    const int qt = blockIdx.x & 63;
    const int h  = (blockIdx.x >> 6) & 7;
    const int b  = blockIdx.x >> 9;
    const int hg = lane >> 4, cn = lane & 15, kb = hg * 16;

    const _Float16* Q = qh  + (((size_t)b * N_HEAD + h) * SEQ + qt * 16) * D_HEAD;
    const _Float16* K = kh  + ((size_t)b * N_HEAD + h) * SEQ * D_HEAD;
    const _Float16* V = vhT + ((size_t)b * N_HEAD + h) * D_HEAD * SEQ;  // [64, L]
    const unsigned char* M = mask + ((size_t)b * SEQ + qt * 16) * SEQ;
    float* ATT = attn_out + (((size_t)b * N_HEAD + h) * SEQ + qt * 16) * SEQ;

    // Q fragments (rows qt*16.., K-dim = 64 -> two frags), pre-scaled by 1/8
    v16h qa0, qa1;
    {
        const _Float16* p = Q + (size_t)cn * D_HEAD + kb;
#pragma unroll
        for (int j = 0; j < 8; ++j) {
            qa0[j]     = (_Float16)((float)p[j]      * 0.125f);
            qa0[8 + j] = (_Float16)((float)p[8 + j] * 0.125f);
            qa1[j]     = (_Float16)((float)p[32 + j] * 0.125f);
            qa1[8 + j] = (_Float16)((float)p[40 + j] * 0.125f);
        }
    }

    // ---- scores: S = (Q/8) @ K^T, masked, into LDS ----
#pragma unroll 1
    for (int jt = 0; jt < 64; ++jt) {
        v8f s = {};
        // B = K^T: element (kdim=d, n=key) at kh[key*64 + d] (contiguous in d)
        const _Float16* pb = K + (size_t)(jt * 16 + cn) * D_HEAD + hg * 16;
        v16h b0, b1;
#pragma unroll
        for (int e = 0; e < 16; ++e) { b0[e] = pb[e]; b1[e] = pb[32 + e]; }
        s = wmma16(qa0, b0, s);
        s = wmma16(qa1, b1, s);
#pragma unroll
        for (int i = 0; i < 8; ++i) {
            int row = i + 8 * hg;
            int key = jt * 16 + cn;
            float val = M[(size_t)row * SEQ + key] ? -1e9f : s[i];
            S[row * SEQ + key] = val;
        }
    }
    __syncthreads();

    // ---- softmax stats: lane (cn,hg) scans half of row cn ----
    const float4* r4 = (const float4*)(S + cn * SEQ + hg * 512);
    float mx = -3.4e38f;
#pragma unroll 4
    for (int i = 0; i < 128; ++i) {
        float4 t = r4[i];
        mx = fmaxf(mx, fmaxf(fmaxf(t.x, t.y), fmaxf(t.z, t.w)));
    }
    mx = fmaxf(mx, __shfl_xor(mx, 16, 32));
    float sum = 0.f;
#pragma unroll 4
    for (int i = 0; i < 128; ++i) {
        float4 t = r4[i];
        sum += __expf(t.x - mx) + __expf(t.y - mx) + __expf(t.z - mx) + __expf(t.w - mx);
    }
    sum += __shfl_xor(sum, 16, 32);
    const float inv = 1.0f / sum;
    __syncthreads();

    // ---- normalize: write f32 attn to HBM, repack f16 probs in-place ----
    _Float16* P16 = (_Float16*)S;
#pragma unroll 1
    for (int rr = 0; rr < 16; ++rr) {
        float m_r = __shfl(mx, rr, 32);
        float i_r = __shfl(inv, rr, 32);
#pragma unroll 4
        for (int i = 0; i < 32; ++i) {
            int idx = i * 32 + lane;
            float sv = S[rr * SEQ + idx];              // f32 read (byte 4*e)
            float pv = __expf(sv - m_r) * i_r;
            ATT[(size_t)rr * SEQ + idx] = pv;          // coalesced f32 store
            P16[rr * SEQ + idx] = (_Float16)pv;        // f16 write (byte 2*e) — only
        }                                              // touches already-consumed bytes
    }
    __syncthreads();

    // ---- O = P @ V ----
    v8f oacc[4] = {};
#pragma unroll 1
    for (int kc = 0; kc < 32; ++kc) {                  // 1024 keys / 32
        const _Float16* pa = P16 + (size_t)cn * SEQ + kc * 32;
        v16h a;
#pragma unroll
        for (int j = 0; j < 8; ++j) {
            a[j]     = pa[kb + j];
            a[8 + j] = pa[8 + kb + j];
        }
#pragma unroll
        for (int dt = 0; dt < 4; ++dt) {
            // B frag from V^T [64, L]: lane n = dt*16+cn, contiguous in key
            const _Float16* pb = V + (size_t)(dt * 16 + cn) * SEQ + kc * 32 + hg * 16;
            v16h bf;
#pragma unroll
            for (int e = 0; e < 16; ++e) bf[e] = pb[e];
            oacc[dt] = wmma16(a, bf, oacc[dt]);
        }
    }

    // store O as f16 [B, L, 512] (heads re-interleaved) for the output GEMM
    const size_t orow0 = (size_t)b * SEQ + qt * 16;
#pragma unroll
    for (int dt = 0; dt < 4; ++dt) {
#pragma unroll
        for (int i = 0; i < 8; ++i) {
            int row = i + 8 * hg;
            int col = h * D_HEAD + dt * 16 + cn;
            o_out[(orow0 + row) * D_MODEL + col] = (_Float16)oacc[dt][i];
        }
    }
}

// ---------------------------------------------------------------------------
// Kernel C: out = LN(O @ Wfc + residual) * gamma + beta.  One wave per 16 rows.
// ---------------------------------------------------------------------------
__global__ __launch_bounds__(32)
void out_ln_kernel(const _Float16* __restrict__ O, const _Float16* __restrict__ WfcT,
                   const float* __restrict__ resid, const float* __restrict__ gamma,
                   const float* __restrict__ beta, float* __restrict__ out) {
    __shared__ float    rowbuf[16 * D_MODEL];          // 32 KB
    __shared__ _Float16 a16[16 * D_MODEL];             // 16 KB

    const int lane = threadIdx.x;
    const int hg = lane >> 4, cn = lane & 15, kb = hg * 16;
    const size_t row0 = (size_t)blockIdx.x * 16;

    // stage the contiguous 16x512 f16 A strip in LDS
    {
        const uint4* src = (const uint4*)(O + row0 * D_MODEL);
        uint4* dst = (uint4*)a16;
        for (int i = lane; i < 1024; i += 32) dst[i] = src[i];
    }
    __syncthreads();

#pragma unroll 1
    for (int nt = 0; nt < 32; ++nt) {
        v8f acc = {};
#pragma unroll 1
        for (int kc = 0; kc < 16; ++kc) {
            const _Float16* pa = a16 + (size_t)cn * D_MODEL + kc * 32;
            v16h a;
#pragma unroll
            for (int j = 0; j < 8; ++j) {
                a[j]     = pa[kb + j];
                a[8 + j] = pa[8 + kb + j];
            }
            // B frag from Wfc^T [N, K]: lane n = nt*16+cn, contiguous in K
            const _Float16* pb = WfcT + (size_t)(nt * 16 + cn) * D_MODEL
                                      + kc * 32 + hg * 16;
            v16h b;
#pragma unroll
            for (int e = 0; e < 16; ++e) b[e] = pb[e];
            acc = wmma16(a, b, acc);
        }
#pragma unroll
        for (int i = 0; i < 8; ++i) {
            int r = i + 8 * hg, c = nt * 16 + cn;
            rowbuf[r * D_MODEL + c] = acc[i] + resid[(row0 + r) * D_MODEL + c];
        }
    }
    __syncthreads();

    // LayerNorm: single-pass sum / sumsq per half-row, combine across halves
    float sum = 0.f, sq = 0.f;
    {
        const float4* p4 = (const float4*)(rowbuf + cn * D_MODEL + hg * 256);
#pragma unroll 4
        for (int i = 0; i < 64; ++i) {
            float4 t = p4[i];
            sum += t.x + t.y + t.z + t.w;
            sq  += t.x * t.x + t.y * t.y + t.z * t.z + t.w * t.w;
        }
    }
    sum += __shfl_xor(sum, 16, 32);
    sq  += __shfl_xor(sq, 16, 32);
    float mean = sum * (1.f / 512.f);
    float var  = sq * (1.f / 512.f) - mean * mean;   // population var (jnp.var)
    float rstd = rsqrtf(var + 1e-6f);

#pragma unroll 1
    for (int rr = 0; rr < 16; ++rr) {
        float mu = __shfl(mean, rr, 32);
        float rs = __shfl(rstd, rr, 32);
#pragma unroll
        for (int i = 0; i < 16; ++i) {
            int idx = i * 32 + lane;
            float x = rowbuf[rr * D_MODEL + idx];
            out[(row0 + rr) * D_MODEL + idx] = (x - mu) * rs * gamma[idx] + beta[idx];
        }
    }
}

// ---------------------------------------------------------------------------
extern "C" void kernel_launch(void* const* d_in, const int* in_sizes, int n_in,
                              void* d_out, int out_size, void* d_ws, size_t ws_size,
                              hipStream_t stream) {
    const float* q    = (const float*)d_in[0];
    const float* k    = (const float*)d_in[1];
    const float* v    = (const float*)d_in[2];
    const unsigned char* mask = (const unsigned char*)d_in[3];  // jax bool = 1 byte
    const float* Wq   = (const float*)d_in[4];
    const float* Wk   = (const float*)d_in[5];
    const float* Wv   = (const float*)d_in[6];
    const float* Wfc  = (const float*)d_in[7];
    const float* gamma = (const float*)d_in[8];
    const float* beta  = (const float*)d_in[9];

    float* out = (float*)d_out;                    // [0, 8.4M): out, then attn

    // workspace (f16): qh, kh, vhT, O (16 MB each) + 4 transposed weights (2 MB)
    _Float16* ws   = (_Float16*)d_ws;
    _Float16* qh   = ws;
    _Float16* kh   = ws + PROJ_ELEMS;
    _Float16* vhT  = ws + 2 * PROJ_ELEMS;
    _Float16* oO   = ws + 3 * PROJ_ELEMS;
    _Float16* WqT  = ws + 4 * PROJ_ELEMS;
    _Float16* WkT  = WqT + W_ELEMS;
    _Float16* WvT  = WkT + W_ELEMS;
    _Float16* WfcT = WvT + W_ELEMS;

    // 0) one-time weight convert+transpose to f16 (L2-resident afterwards)
    prep_weights_kernel<<<4096, 256, 0, stream>>>(Wq, Wk, Wv, Wfc,
                                                  WqT, WkT, WvT, WfcT);

    // A) Q/K/V projections: 3 * 1024 * 8 waves, 8 waves/block
    proj_gemm_kernel<<<3072, 256, 0, stream>>>(q, k, v, WqT, WkT, WvT,
                                               qh, kh, vhT);

    // B) attention: B*H*(L/16) = 8192 single-wave blocks, 64 KB LDS each
    attn_kernel<<<8192, 32, 0, stream>>>(qh, kh, vhT, mask, out + OUT_ELEMS, oO);

    // C) output projection + residual + LayerNorm: 16384/16 single-wave blocks
    out_ln_kernel<<<1024, 32, 0, stream>>>(oO, WfcT, q, gamma, beta, out);
}
